// RNNModel_2035814498381
// MI455X (gfx1250) — matrix operand order repaired and checked by reading, not compiled, not run on hardware
//
#include <hip/hip_runtime.h>

typedef __attribute__((ext_vector_type(16))) __bf16        v16bf;
typedef __attribute__((ext_vector_type(8)))  float         v8f;
typedef __attribute__((ext_vector_type(4)))  unsigned int  u32x4;
typedef __attribute__((ext_vector_type(4)))  float         f32x4;

union Frag32B {
  v16bf v;
  u32x4 q[2];
};

#define RNN_B 512
#define RNN_T 256
#define RNN_D 512
#define RNN_H 1024
#define BM    16
#define NTHREADS 512          // 16 waves
#define XSTRIDE 520           // 512 + 8: 16B-aligned rows, 4-dword bank skew
#define HSTRIDE 1032          // 1024 + 8: 16B-aligned rows, 4-dword bank skew

__device__ __forceinline__ unsigned short f32_to_bf16(float f) {
  unsigned int u = __float_as_uint(f);
  u += 0x7FFFu + ((u >> 16) & 1u);        // round-to-nearest-even
  return (unsigned short)(u >> 16);
}
__device__ __forceinline__ float bf16_to_f32(unsigned short h) {
  return __uint_as_float(((unsigned int)h) << 16);
}

// gfx1250 native packed bf16 convert: dst = {bf16(hi), bf16(lo)}
__device__ __forceinline__ unsigned int pk_bf16(float lo, float hi) {
  unsigned int d;
  asm("v_cvt_pk_bf16_f32 %0, %1, %2" : "=v"(d) : "v"(lo), "v"(hi));
  return d;
}
// gfx1250 native transcendental tanh (TRANS op, co-executes with XDL WMMA)
__device__ __forceinline__ float fast_tanh(float a) {
  float r;
  asm("v_tanh_f32 %0, %1" : "=v"(r) : "v"(a));
  return r;
}

// ---------------------------------------------------------------------------
// Pre-pass: convert W (512x1024) and U (1024x1024) fp32 -> bf16, pre-swizzled
// into a 32x16 B-fragment layout in contiguous k order (same order as the A loads):
//   fragment(kb, nt) = 512 bf16, packed as [lane 0..31][elem 0..15]
//   lane L, elem e -> K = kb*32 + 16*(L>=16) + e, N = nt*16 + (L&15)
// so the hot loop loads each fragment with two coalesced global_load_b128.
// ---------------------------------------------------------------------------
__global__ __launch_bounds__(256) void pack_weights_kernel(
    const float* __restrict__ W, const float* __restrict__ U,
    unsigned short* __restrict__ wp, unsigned short* __restrict__ up)
{
  long idx = (long)blockIdx.x * 256 + threadIdx.x;
  const long WE = (long)16 * 64 * 512;   // W fragments: 16 kb x 64 nt
  const long UE = (long)32 * 64 * 512;   // U fragments: 32 kb x 64 nt
  if (idx < WE) {
    long frag = idx >> 9;
    int  r  = (int)(idx & 511);
    int  L  = r >> 4, e = r & 15;
    int  kb = (int)(frag >> 6), nt = (int)(frag & 63);
    int  k  = kb * 32 + ((L >= 16) ? 16 : 0) + e;
    int  n  = nt * 16 + (L & 15);
    wp[idx] = f32_to_bf16(W[(long)k * RNN_H + n]);
  } else if (idx < WE + UE) {
    long i2 = idx - WE;
    long frag = i2 >> 9;
    int  r  = (int)(i2 & 511);
    int  L  = r >> 4, e = r & 15;
    int  kb = (int)(frag >> 6), nt = (int)(frag & 63);
    int  k  = kb * 32 + ((L >= 16) ? 16 : 0) + e;
    int  n  = nt * 16 + (L & 15);
    up[i2] = f32_to_bf16(U[(long)k * RNN_H + n]);
  }
}

// Stage helpers: each thread owns 16 consecutive columns of one row.
__device__ __forceinline__ void load_x_tile(const float* __restrict__ x,
                                            int bb, int row, int col0, int t,
                                            f32x4 r[4]) {
  long gi = ((long)(bb + row) * RNN_T + t) * RNN_D + col0;
  const f32x4* p = (const f32x4*)(x + gi);
  r[0] = __builtin_nontemporal_load(p + 0);   // streamed once: NT hint
  r[1] = __builtin_nontemporal_load(p + 1);
  r[2] = __builtin_nontemporal_load(p + 2);
  r[3] = __builtin_nontemporal_load(p + 3);
}
__device__ __forceinline__ void store_x_tile(unsigned short* __restrict__ xrow,
                                             int col0, const f32x4 r[4]) {
  u32x4 p0, p1;
  p0.x = pk_bf16(r[0].x, r[0].y);  p0.y = pk_bf16(r[0].z, r[0].w);
  p0.z = pk_bf16(r[1].x, r[1].y);  p0.w = pk_bf16(r[1].z, r[1].w);
  p1.x = pk_bf16(r[2].x, r[2].y);  p1.y = pk_bf16(r[2].z, r[2].w);
  p1.z = pk_bf16(r[3].x, r[3].y);  p1.w = pk_bf16(r[3].z, r[3].w);
  *(u32x4*)(xrow + col0)     = p0;
  *(u32x4*)(xrow + col0 + 8) = p1;
}

// ---------------------------------------------------------------------------
// Fused RNN: each workgroup owns 16 batch rows, iterates all T=256 steps with
// h resident in LDS (bf16). Per step: acc = bias; acc += x_t@W (K=512);
// acc += h@U (K=1024); h = tanh(acc). x_{t+1} is software-pipelined: loaded
// into registers at step start (latency hidden behind 48 WMMA k-blocks),
// converted/stored to LDS after the mid-barrier. 16 waves x 64 columns each.
// ---------------------------------------------------------------------------
__global__ __launch_bounds__(NTHREADS) void rnn_fused_kernel(
    const float* __restrict__ x,  const float* __restrict__ b,
    const float* __restrict__ Wd, const float* __restrict__ bd,
    const unsigned short* __restrict__ wp, const unsigned short* __restrict__ up,
    float* __restrict__ out)
{
  __shared__ unsigned short hsh[BM][HSTRIDE];
  __shared__ unsigned short xsh[BM][XSTRIDE];

  const int tid    = threadIdx.x;
  const int wave   = tid >> 5;
  const int lane   = tid & 31;
  const int laneLo = lane & 15;
  const int hi     = lane >> 4;          // 0: K-low half, 1: K-high half
  const int bb     = blockIdx.x * BM;    // first batch row of this block
  const int nt0    = wave * 4;           // first of 4 n-tiles for this wave

  const int srow = tid >> 5;             // staging: row (== wave index)
  const int scol = (tid & 31) * 16;      // staging: 16 consecutive columns

  // h_0 = 0
  for (int i = tid; i < BM * HSTRIDE; i += NTHREADS)
    ((unsigned short*)hsh)[i] = 0;

  // Bias preload per C-fragment column (C layout: N = lane&15 for all VGPRs).
  float biasv[4];
#pragma unroll
  for (int j = 0; j < 4; ++j)
    biasv[j] = b[nt0 * 16 + j * 16 + laneLo];

  // Prologue: stage x_0 into LDS.
  f32x4 xr[4];
  load_x_tile(x, bb, srow, scol, 0, xr);
  store_x_tile(&xsh[srow][0], scol, xr);
  __syncthreads();

  for (int t = 0; t < RNN_T; ++t) {
    // Issue next step's x loads now; they complete during the WMMA section.
    int tn = (t + 1 < RNN_T) ? t + 1 : t;
    load_x_tile(x, bb, srow, scol, tn, xr);

    v8f acc[4];
#pragma unroll
    for (int j = 0; j < 4; ++j)
#pragma unroll
      for (int r = 0; r < 8; ++r)
        acc[j][r] = biasv[j];

    // ---- Part 1: x_t @ W (K = 512) ----
#pragma unroll 2
    for (int kb = 0; kb < RNN_D / 32; ++kb) {
      Frag32B a;
      const unsigned short* ap = &xsh[laneLo][kb * 32 + 16 * hi];
      a.q[0] = *(const u32x4*)(ap);        // K 0..7  (of this half)
      a.q[1] = *(const u32x4*)(ap + 8);    // K 8..15 (of this half)
      const unsigned short* fb = wp + ((long)kb * 64 + nt0) * 512 + lane * 16;
      __builtin_prefetch(fb + 64 * 512, 0, 1);   // next kb's fragments
#pragma unroll
      for (int j = 0; j < 4; ++j) {
        Frag32B bfr;
        const u32x4* bp = (const u32x4*)(fb + j * 512);
        bfr.q[0] = bp[0];
        bfr.q[1] = bp[1];
        acc[j] = __builtin_amdgcn_wmma_f32_16x16x32_bf16(
            false, a.v, false, bfr.v, (short)0, acc[j], false, false);
      }
    }
    // ---- Part 2: h_{t-1} @ U (K = 1024) ----
#pragma unroll 2
    for (int kb = 0; kb < RNN_H / 32; ++kb) {
      Frag32B a;
      const unsigned short* ap = &hsh[laneLo][kb * 32 + 16 * hi];
      a.q[0] = *(const u32x4*)(ap);
      a.q[1] = *(const u32x4*)(ap + 8);
      const unsigned short* fb = up + ((long)kb * 64 + nt0) * 512 + lane * 16;
      __builtin_prefetch(fb + 64 * 512, 0, 1);
#pragma unroll
      for (int j = 0; j < 4; ++j) {
        Frag32B bfr;
        const u32x4* bp = (const u32x4*)(fb + j * 512);
        bfr.q[0] = bp[0];
        bfr.q[1] = bp[1];
        acc[j] = __builtin_amdgcn_wmma_f32_16x16x32_bf16(
            false, a.v, false, bfr.v, (short)0, acc[j], false, false);
      }
    }
    __syncthreads();   // all waves done reading xsh (x_t) and hsh (h_{t-1})

    // Commit the pipelined x_{t+1} tile to LDS.
    store_x_tile(&xsh[srow][0], scol, xr);

    // h_t = tanh(acc) -> LDS (C layout: M = r + 8*hi, N = tile + lane&15)
#pragma unroll
    for (int j = 0; j < 4; ++j) {
      int n = nt0 * 16 + j * 16 + laneLo;
#pragma unroll
      for (int r = 0; r < 8; ++r) {
        int m = r + 8 * hi;
        hsh[m][n] = (unsigned short)pk_bf16(fast_tanh(acc[j][r]), 0.f);
      }
    }
    __syncthreads();   // h_t (and x_{t+1}) fully written before next step
  }

  // Final Dense(1): logits[b] = h_T[b,:] @ Wd + bd. Reuse xsh as fp32 scratch.
  {
    float* red = (float*)xsh;        // 16 rows x 32 segments = 2 KB
    int row = tid >> 5;              // 32 threads per row
    int seg = tid & 31;
    float s = 0.f;
    for (int c = seg * 32; c < seg * 32 + 32; ++c)
      s += bf16_to_f32(hsh[row][c]) * Wd[c];
    red[row * 32 + seg] = s;
    __syncthreads();
    if (tid < BM) {
      float s2 = bd[0];
#pragma unroll
      for (int k = 0; k < 32; ++k) s2 += red[tid * 32 + k];
      out[bb + tid] = s2;
    }
  }
}

extern "C" void kernel_launch(void* const* d_in, const int* in_sizes, int n_in,
                              void* d_out, int out_size, void* d_ws, size_t ws_size,
                              hipStream_t stream) {
  const float* x  = (const float*)d_in[0];   // [512,256,512]
  const float* W  = (const float*)d_in[1];   // [512,1024]
  const float* U  = (const float*)d_in[2];   // [1024,1024]
  const float* b  = (const float*)d_in[3];   // [1024]
  const float* Wd = (const float*)d_in[4];   // [1024,1]
  const float* bd = (const float*)d_in[5];   // [1]
  float* out = (float*)d_out;                // [512,1]

  unsigned short* wp = (unsigned short*)d_ws;        // 512K bf16 (1 MB)
  unsigned short* up = wp + (long)16 * 64 * 512;     // 1M bf16   (2 MB)

  const long packElems = (long)16 * 64 * 512 + (long)32 * 64 * 512;
  int packBlocks = (int)((packElems + 255) / 256);
  pack_weights_kernel<<<packBlocks, 256, 0, stream>>>(W, U, wp, up);

  rnn_fused_kernel<<<RNN_B / BM, NTHREADS, 0, stream>>>(x, b, Wd, bd, wp, up, out);
}
